// GatedModel_78048145703097
// MI455X (gfx1250) — matrix operand order repaired and checked by reading, not compiled, not run on hardware
//
#include <hip/hip_runtime.h>
#include <math.h>

// ---------------------------------------------------------------------------
// CDNA5 (gfx1250) WMMA types
// ---------------------------------------------------------------------------
typedef __attribute__((ext_vector_type(16))) _Float16 v16h;
typedef __attribute__((ext_vector_type(8)))  float    v8f;
typedef __attribute__((ext_vector_type(4)))  unsigned int v4u;

union Frag16 { v16h h; v4u u[2]; };

__device__ inline float wave_sum(float v) {
  #pragma unroll
  for (int o = 16; o; o >>= 1) v += __shfl_xor(v, o, 32);
  return v;
}

__device__ inline float gelu_exact(float x) {
  return 0.5f * x * (1.0f + erff(x * 0.70710678118654752f));
}

// ---------------------------------------------------------------------------
// Generic GEMM:  Y[M x N] = X[M x K] * W[N x K]^T + bias,  f32 in/out,
// f16 multiply + f32 accumulate (v_wmma_f32_16x16x32_f16).
// 4 waves / block; wave w owns D-tile (blockIdx.x*4+w, blockIdx.y).
// K-chunk = 64 (two WMMAs per barrier pair).  Both fast and slow staging
// paths issue batched unconditional loads (clamped addresses); bounds are
// applied afterwards with arithmetic float masks (no exec/scalar branches
// around loads -> no per-element s_wait_loadcnt).
// ---------------------------------------------------------------------------
__global__ __launch_bounds__(128) void gemm_bias_kernel(
    const float* __restrict__ X, int ldx,
    const float* __restrict__ Wm, int ldw,
    const float* __restrict__ bias,
    float* __restrict__ Y, int ldy,
    int M, int N, int K)
{
  __shared__ __align__(16) _Float16 As[4][1024];  // 16 rows x 64 k, per wave
  __shared__ __align__(16) _Float16 Bs[1024];     // 16 cols x 64 k, shared

  const int tid  = threadIdx.x;
  const int lane = tid & 31;
  const int wv   = tid >> 5;
  const int row0 = (blockIdx.x * 4 + wv) * 16;
  const int col0 = blockIdx.y * 16;

  const bool colFull = (col0 + 16 <= N);
  const bool rowFull = (row0 + 16 <= M);

  v8f acc = {0.f, 0.f, 0.f, 0.f, 0.f, 0.f, 0.f, 0.f};

  for (int k0 = 0; k0 < K; k0 += 64) {
    const bool kFull = (k0 + 64 <= K);

    // ---- stage B tile [16][64] (block-shared)
    if (kFull && colFull) {
      float bv[8];
      #pragma unroll
      for (int t = 0; t < 8; t++) {
        int e = t * 128 + tid;
        int n = e >> 6, c = e & 63;
        bv[t] = Wm[(size_t)(col0 + n) * ldw + k0 + c];
      }
      if (k0 + 64 < K)
        __builtin_prefetch(&Wm[(size_t)col0 * ldw + k0 + 64], 0, 1);
      #pragma unroll
      for (int t = 0; t < 8; t++) Bs[t * 128 + tid] = (_Float16)bv[t];
    } else {
      float bv[8];
      #pragma unroll
      for (int t = 0; t < 8; t++) {           // batched clamped loads
        int e = t * 128 + tid;
        int n = e >> 6, c = e & 63;
        int wr = col0 + n, kk = k0 + c;
        int wrc = wr < N ? wr : N - 1;
        int kkc = kk < K ? kk : K - 1;
        bv[t] = Wm[(size_t)wrc * ldw + kkc];
      }
      #pragma unroll
      for (int t = 0; t < 8; t++) {           // arithmetic mask, no branches
        int e = t * 128 + tid;
        int n = e >> 6, c = e & 63;
        float msk = (float)((int)(col0 + n < N) & (int)(k0 + c < K));
        Bs[e] = (_Float16)(bv[t] * msk);
      }
    }

    // ---- stage A tile [16][64] (per wave)
    if (kFull && rowFull) {
      const float* xp = X + (size_t)row0 * ldx + k0 + lane;
      float av[32];
      #pragma unroll
      for (int t = 0; t < 32; t++) {
        int r = t >> 1, c = (t & 1) << 5;
        av[t] = xp[(size_t)r * ldx + c];
      }
      if (k0 + 64 < K)
        __builtin_prefetch(&X[(size_t)row0 * ldx + k0 + 64], 0, 1);
      #pragma unroll
      for (int t = 0; t < 32; t++) As[wv][t * 32 + lane] = (_Float16)av[t];
    } else {
      float av[32];
      #pragma unroll
      for (int t = 0; t < 32; t++) {          // batched clamped loads
        int r = t >> 1, c = ((t & 1) << 5) + lane;
        int gr = row0 + r, kk = k0 + c;
        int grc = gr < M ? gr : M - 1;
        int kkc = kk < K ? kk : K - 1;
        av[t] = X[(size_t)grc * ldx + kkc];
      }
      #pragma unroll
      for (int t = 0; t < 32; t++) {          // arithmetic mask, no branches
        int r = t >> 1, c = ((t & 1) << 5) + lane;
        float msk = (float)((int)(row0 + r < M) & (int)(k0 + c < K));
        As[wv][t * 32 + lane] = (_Float16)(av[t] * msk);
      }
    }
    __syncthreads();

    // ---- two 16x16x32 WMMAs over the 64-wide chunk
    const int g  = lane >> 4;
    const int mr = lane & 15;
    #pragma unroll
    for (int s = 0; s < 2; s++) {
      Frag16 a, b;
      a.u[0] = *(const v4u*)&As[wv][mr * 64 + 32 * s + 8 * g];
      a.u[1] = *(const v4u*)&As[wv][mr * 64 + 32 * s + 16 + 8 * g];
      b.u[0] = *(const v4u*)&Bs[mr * 64 + 32 * s + 8 * g];
      b.u[1] = *(const v4u*)&Bs[mr * 64 + 32 * s + 16 + 8 * g];
      acc = __builtin_amdgcn_wmma_f32_16x16x32_f16(
          false, a.h, false, b.h, (short)0, acc, false, false);
    }
    __syncthreads();
  }

  // ---- epilogue: D layout (M = r + 8g, N = lane&15)
  const int g   = lane >> 4;
  const int col = col0 + (lane & 15);
  if (col < N) {
    float bb = bias ? bias[col] : 0.0f;
    #pragma unroll
    for (int r = 0; r < 8; r++) {
      int row = row0 + r + 8 * g;
      if (row < M) Y[(size_t)row * ldy + col] = acc[r] + bb;
    }
  }
}

// ---------------------------------------------------------------------------
// 3x3 conv as implicit GEMM, NCHW in/out.  Compile-time spatial dims so the
// per-element im2col index math strength-reduces (no runtime idiv).
// M = B*HOUT*WOUT, N = Cout, K = Cin*9.  Pad handled by clamp + float mask;
// loads batched in their own loop (no predication).
// ---------------------------------------------------------------------------
template <int HIN, int WIN, int HOUT, int WOUT, int PAD>
__global__ __launch_bounds__(128) void conv3x3_wmma_kernel(
    const float* __restrict__ X, const float* __restrict__ Wm,
    const float* __restrict__ bias, float* __restrict__ Y,
    int Bn, int Cin, int Cout)
{
  __shared__ __align__(16) _Float16 As[4][1024];
  __shared__ __align__(16) _Float16 Bs[1024];

  constexpr int HW = HOUT * WOUT;
  const int M = Bn * HW;
  const int N = Cout;
  const int K = Cin * 9;

  const int tid  = threadIdx.x;
  const int lane = tid & 31;
  const int wv   = tid >> 5;
  const int row0 = (blockIdx.x * 4 + wv) * 16;
  const int col0 = blockIdx.y * 16;
  const bool colFull = (col0 + 16 <= N);

  v8f acc = {0.f, 0.f, 0.f, 0.f, 0.f, 0.f, 0.f, 0.f};

  for (int k0 = 0; k0 < K; k0 += 64) {
    const bool kFull = (k0 + 64 <= K);

    // ---- B tile: weights are [Cout][Cin*9] row-major
    {
      float bv[8];
      if (kFull && colFull) {
        #pragma unroll
        for (int t = 0; t < 8; t++) {
          int e = t * 128 + tid;
          int n = e >> 6, c = e & 63;
          bv[t] = Wm[(size_t)(col0 + n) * K + k0 + c];
        }
        if (k0 + 64 < K)
          __builtin_prefetch(&Wm[(size_t)col0 * K + k0 + 64], 0, 1);
        #pragma unroll
        for (int t = 0; t < 8; t++) Bs[t * 128 + tid] = (_Float16)bv[t];
      } else {
        #pragma unroll
        for (int t = 0; t < 8; t++) {
          int e = t * 128 + tid;
          int n = e >> 6, c = e & 63;
          int wr = col0 + n, kk = k0 + c;
          int wrc = wr < N ? wr : N - 1;
          int kkc = kk < K ? kk : K - 1;
          bv[t] = Wm[(size_t)wrc * K + kkc];
        }
        #pragma unroll
        for (int t = 0; t < 8; t++) {
          int e = t * 128 + tid;
          int n = e >> 6, c = e & 63;
          float msk = (float)((int)(col0 + n < N) & (int)(k0 + c < K));
          Bs[e] = (_Float16)(bv[t] * msk);
        }
      }
    }

    // ---- A tile: on-the-fly im2col gather; batched clamped loads, then mask
    {
      float av[32];
      #pragma unroll
      for (int t = 0; t < 32; t++) {
        int r = t >> 1, c = ((t & 1) << 5) + lane;
        int gr = row0 + r, kk = k0 + c;
        int grc = gr < M ? gr : M - 1;
        int kkc = kk < K ? kk : K - 1;
        int bb = grc / HW, pos = grc % HW;     // HW is constexpr
        int oy = pos / WOUT, ox = pos % WOUT;
        int ci = kkc / 9, rr = kkc % 9;
        int ky = rr / 3, kx = rr % 3;
        int iy = oy + ky - PAD, ix = ox + kx - PAD;
        int iyc = iy < 0 ? 0 : (iy >= HIN ? HIN - 1 : iy);
        int ixc = ix < 0 ? 0 : (ix >= WIN ? WIN - 1 : ix);
        av[t] = X[(((size_t)bb * Cin + ci) * HIN + iyc) * WIN + ixc];
      }
      #pragma unroll
      for (int t = 0; t < 32; t++) {
        int r = t >> 1, c = ((t & 1) << 5) + lane;
        int gr = row0 + r, kk = k0 + c;
        int grc = gr < M ? gr : M - 1;
        int kkc = kk < K ? kk : K - 1;
        int pos = grc % HW;
        int oy = pos / WOUT, ox = pos % WOUT;
        int rr = kkc % 9;
        int ky = rr / 3, kx = rr % 3;
        int iy = oy + ky - PAD, ix = ox + kx - PAD;
        int ok = (int)(gr < M) & (int)(kk < K) &
                 (int)(iy >= 0) & (int)(iy < HIN) &
                 (int)(ix >= 0) & (int)(ix < WIN);
        As[wv][t * 32 + lane] = (_Float16)(av[t] * (float)ok);
      }
    }
    __syncthreads();

    const int g  = lane >> 4;
    const int mr = lane & 15;
    #pragma unroll
    for (int s = 0; s < 2; s++) {
      Frag16 a, b;
      a.u[0] = *(const v4u*)&As[wv][mr * 64 + 32 * s + 8 * g];
      a.u[1] = *(const v4u*)&As[wv][mr * 64 + 32 * s + 16 + 8 * g];
      b.u[0] = *(const v4u*)&Bs[mr * 64 + 32 * s + 8 * g];
      b.u[1] = *(const v4u*)&Bs[mr * 64 + 32 * s + 16 + 8 * g];
      acc = __builtin_amdgcn_wmma_f32_16x16x32_f16(
          false, a.h, false, b.h, (short)0, acc, false, false);
    }
    __syncthreads();
  }

  const int g   = lane >> 4;
  const int col = col0 + (lane & 15);
  if (col < N) {
    float bb = bias[col];
    #pragma unroll
    for (int r = 0; r < 8; r++) {
      int row = row0 + r + 8 * g;
      if (row < M) {
        int bidx = row / HW, pos = row % HW;
        Y[((size_t)bidx * Cout + col) * HW + pos] = acc[r] + bb;  // NCHW
      }
    }
  }
}

// ---------------------------------------------------------------------------
// LayerNorm over channels, NCHW input -> row-major tokens [B*N, C]
// ---------------------------------------------------------------------------
__global__ void ln_spatial_kernel(const float* __restrict__ X,
                                  const float* __restrict__ g,
                                  const float* __restrict__ bt,
                                  float* __restrict__ R,
                                  int Bn, int C, int N)
{
  int gw   = (blockIdx.x * blockDim.x + threadIdx.x) >> 5;
  int lane = threadIdx.x & 31;
  if (gw >= Bn * N) return;
  int b = gw / N, pos = gw % N;
  const float* xb = X + (size_t)b * C * N + pos;
  float s = 0.f, ss = 0.f;
  for (int c = lane; c < C; c += 32) {
    float v = xb[(size_t)c * N];
    s += v; ss += v * v;
  }
  s = wave_sum(s); ss = wave_sum(ss);
  float m = s / C, var = ss / C - m * m;
  float rs = rsqrtf(var + 1e-6f);
  for (int c = lane; c < C; c += 32) {
    float v = xb[(size_t)c * N];
    R[(size_t)gw * C + c] = (v - m) * rs * g[c] + bt[c];
  }
}

// "Spectral" LN: t = transpose(x,(0,2,3,1)).reshape(B,C,H*W), LN over H*W.
__global__ void ln_spectral_kernel(const float* __restrict__ X,
                                   const float* __restrict__ g,
                                   const float* __restrict__ bt,
                                   float* __restrict__ R,
                                   int Bn, int C, int H, int W)
{
  int HW   = H * W;
  int gw   = (blockIdx.x * blockDim.x + threadIdx.x) >> 5;
  int lane = threadIdx.x & 31;
  if (gw >= Bn * C) return;
  int b = gw / C, cp = gw % C;
  float s = 0.f, ss = 0.f;
  for (int n = lane; n < HW; n += 32) {
    int idx = cp * HW + n;
    int h = idx / (W * C), w = (idx / C) % W, c = idx % C;
    float v = X[(((size_t)b * C + c) * H + h) * W + w];
    s += v; ss += v * v;
  }
  s = wave_sum(s); ss = wave_sum(ss);
  float m = s / HW, var = ss / HW - m * m;
  float rs = rsqrtf(var + 1e-6f);
  for (int n = lane; n < HW; n += 32) {
    int idx = cp * HW + n;
    int h = idx / (W * C), w = (idx / C) % W, c = idx % C;
    float v = X[(((size_t)b * C + c) * H + h) * W + w];
    R[(size_t)gw * HW + n] = (v - m) * rs * g[n] + bt[n];
  }
}

// ---------------------------------------------------------------------------
// Central attention: logits[n,m] = dw[n] * (q_center . k_m) * hid^-0.5
// ---------------------------------------------------------------------------
__global__ void central_attn_kernel(const float* __restrict__ Q,
                                    const float* __restrict__ K,
                                    const float* __restrict__ V,
                                    float* __restrict__ O,
                                    int N, int hid, int H, int W)
{
  __shared__ float s[64];
  int b = blockIdx.x;
  int lane = threadIdx.x & 31, wv = threadIdx.x >> 5;
  int nw = blockDim.x >> 5;
  int center = (H / 2) * W + (W / 2);
  const float* qc = Q + ((size_t)b * N + center) * hid;
  float scale = rsqrtf((float)hid);

  for (int m = wv; m < N; m += nw) {
    float d = 0.f;
    for (int c = lane; c < hid; c += 32)
      d += qc[c] * K[((size_t)b * N + m) * hid + c];
    d = wave_sum(d);
    if (lane == 0) s[m] = d * scale;
  }
  __syncthreads();

  for (int n = wv; n < N; n += nw) {
    int dy = n / W - H / 2, dx = n % W - W / 2;
    float dwn = expf(1.0f - sqrtf((float)(dy * dy + dx * dx)));
    float maxv = -3.4e38f;
    for (int m = 0; m < N; m++) maxv = fmaxf(maxv, dwn * s[m]);
    float den = 0.f;
    float acc[11];
    #pragma unroll
    for (int t = 0; t < 11; t++) acc[t] = 0.f;
    for (int m = 0; m < N; m++) {
      float p = expf(dwn * s[m] - maxv);
      den += p;
      const float* vm = V + ((size_t)b * N + m) * hid;
      for (int t = 0, c = lane; c < hid; t++, c += 32) acc[t] += p * vm[c];
    }
    float inv = 1.0f / den;
    for (int t = 0, c = lane; c < hid; t++, c += 32)
      O[((size_t)b * N + n) * hid + c] = acc[t] * inv;
  }
}

// ---------------------------------------------------------------------------
// Standard self-attention over T tokens, dim hid (T<=128, hid<=130).
// ---------------------------------------------------------------------------
__global__ void std_attn_kernel(const float* __restrict__ Q,
                                const float* __restrict__ K,
                                const float* __restrict__ V,
                                float* __restrict__ O,
                                int T, int hid)
{
  __shared__ __align__(16) float Ksh[8448];   // max(128*66, 64*130)
  __shared__ float Lrow[4][128];
  int b = blockIdx.x;
  int tid = threadIdx.x, lane = tid & 31, wv = tid >> 5;
  int tot = T * hid;
  for (int i = tid; i < tot; i += blockDim.x)
    Ksh[i] = K[(size_t)b * tot + i];
  __syncthreads();

  float scale = rsqrtf((float)hid);
  for (int n = wv; n < T; n += 4) {
    const float* qn = Q + ((size_t)b * T + n) * hid;
    float qreg[5];
    #pragma unroll
    for (int t = 0; t < 5; t++) {
      int c = lane + 32 * t;
      qreg[t] = (c < hid) ? qn[c] : 0.f;
    }
    float maxv = -3.4e38f;
    for (int m = 0; m < T; m++) {
      float d = 0.f;
      #pragma unroll
      for (int t = 0; t < 5; t++) {
        int c = lane + 32 * t;
        if (c < hid) d += qreg[t] * Ksh[m * hid + c];
      }
      d = wave_sum(d) * scale;
      Lrow[wv][m] = d;
      maxv = fmaxf(maxv, d);
    }
    float den = 0.f;
    for (int m = 0; m < T; m++) {
      float p = expf(Lrow[wv][m] - maxv);
      Lrow[wv][m] = p;
      den += p;
    }
    float inv = 1.0f / den;
    float acc[5] = {0.f, 0.f, 0.f, 0.f, 0.f};
    for (int m = 0; m < T; m++) {
      float p = Lrow[wv][m];
      const float* vm = V + ((size_t)b * T + m) * hid;
      #pragma unroll
      for (int t = 0; t < 5; t++) {
        int c = lane + 32 * t;
        if (c < hid) acc[t] += p * vm[c];
      }
    }
    #pragma unroll
    for (int t = 0; t < 5; t++) {
      int c = lane + 32 * t;
      if (c < hid) O[((size_t)b * T + n) * hid + c] = acc[t] * inv;
    }
  }
}

// ---------------------------------------------------------------------------
// Elementwise pieces
// ---------------------------------------------------------------------------
__global__ void gelu_mul_kernel(const float* __restrict__ Hb,
                                const float* __restrict__ P,
                                float* __restrict__ Tout,
                                size_t rows, int hid)
{
  size_t i = (size_t)blockIdx.x * blockDim.x + threadIdx.x;
  size_t tot = rows * (size_t)hid;
  if (i >= tot) return;
  size_t r = i / hid;
  int c = (int)(i % hid);
  float g = Hb[r * (size_t)(2 * hid) + c];
  Tout[i] = gelu_exact(g) * P[i];
}

__global__ void resid_spatial_kernel(const float* __restrict__ Yr,
                                     const float* __restrict__ Xres,
                                     float* __restrict__ Out,
                                     int Bn, int C, int N)
{
  size_t i = (size_t)blockIdx.x * blockDim.x + threadIdx.x;
  size_t tot = (size_t)Bn * C * N;
  if (i >= tot) return;
  int pos = (int)(i % N);
  size_t t = i / N;
  int c = (int)(t % C);
  size_t b = t / C;
  Out[i] = Yr[((size_t)b * N + pos) * C + c] + Xres[i];
}

__global__ void resid_spectral_kernel(const float* __restrict__ Yr,
                                      const float* __restrict__ Xres,
                                      float* __restrict__ Out,
                                      int Bn, int C, int H, int W)
{
  int HW = H * W;
  size_t i = (size_t)blockIdx.x * blockDim.x + threadIdx.x;
  size_t tot = (size_t)Bn * C * HW;
  if (i >= tot) return;
  int p = (int)(i % HW);
  size_t t = i / HW;
  int c = (int)(t % C);
  size_t b = t / C;
  int idx = p * C + c;
  int cp = idx / HW, n = idx % HW;
  Out[i] = Yr[((size_t)b * C + cp) * HW + n] + Xres[i];
}

__global__ void exchange_kernel(float* __restrict__ x1, float* __restrict__ x2,
                                size_t total, int C, int H, int W, int sp, int ch)
{
  size_t i = (size_t)blockIdx.x * blockDim.x + threadIdx.x;
  if (i >= total) return;
  int w = (int)(i % W);
  int c = (int)((i / ((size_t)H * W)) % C);
  bool cs = sp && ((w & 1) == 0);
  bool cc = ch && ((c & 1) == 0);
  bool cond = (cs != cc);
  float a = x1[i], b = x2[i];
  x1[i] = cond ? b : a;
  x2[i] = cond ? a : b;
}

// ---------------------------------------------------------------------------
// Host-side orchestration
// ---------------------------------------------------------------------------
struct GP {
  const float *kb, *kw, *pb, *pw, *qb, *qw, *vb, *vw;
  const float *f1b, *f1w, *f2b, *f2w, *nb, *ng;
};

// pytree flatten order (dict keys sorted): attn{k_b,k_w,proj_b,proj_w,q_b,q_w,
// v_b,v_w}, fc1_b, fc1_w, fc2_b, fc2_w, norm_b, norm_g
static GP mkGP(void* const* din, int base) {
  GP p;
  p.kb  = (const float*)din[base + 0];  p.kw  = (const float*)din[base + 1];
  p.pb  = (const float*)din[base + 2];  p.pw  = (const float*)din[base + 3];
  p.qb  = (const float*)din[base + 4];  p.qw  = (const float*)din[base + 5];
  p.vb  = (const float*)din[base + 6];  p.vw  = (const float*)din[base + 7];
  p.f1b = (const float*)din[base + 8];  p.f1w = (const float*)din[base + 9];
  p.f2b = (const float*)din[base + 10]; p.f2w = (const float*)din[base + 11];
  p.nb  = (const float*)din[base + 12]; p.ng  = (const float*)din[base + 13];
  return p;
}

static void gemm(const float* X, int ldx, const float* Wm, int ldw,
                 const float* b, float* Y, int ldy,
                 int M, int N, int K, hipStream_t s)
{
  dim3 g((M + 63) / 64, (N + 15) / 16);
  gemm_bias_kernel<<<g, 128, 0, s>>>(X, ldx, Wm, ldw, b, Y, ldy, M, N, K);
}

static void gated_spatial(const float* x, float* y, int Bn, int C, int H, int W,
                          int hid, const GP& p, float* arena, hipStream_t s)
{
  int N = H * W;
  size_t rows = (size_t)Bn * N;
  float* R  = arena;
  float* Hb = R + rows * C;
  float* Qb = Hb + rows * (size_t)(2 * hid);
  float* Kb = Qb + rows * hid;
  float* Vb = Kb + rows * hid;
  float* Ob = Vb + rows * hid;

  ln_spatial_kernel<<<(rows * 32 + 255) / 256, 256, 0, s>>>(
      x, p.ng, p.nb, R, Bn, C, N);
  gemm(R, C, p.f1w, C, p.f1b, Hb, 2 * hid, (int)rows, 2 * hid, C, s);
  gemm(Hb + hid, 2 * hid, p.qw, hid, p.qb, Qb, hid, (int)rows, hid, hid, s);
  gemm(Hb + hid, 2 * hid, p.kw, hid, p.kb, Kb, hid, (int)rows, hid, hid, s);
  gemm(Hb + hid, 2 * hid, p.vw, hid, p.vb, Vb, hid, (int)rows, hid, hid, s);
  central_attn_kernel<<<Bn, 128, 0, s>>>(Qb, Kb, Vb, Ob, N, hid, H, W);
  gemm(Ob, hid, p.pw, hid, p.pb, Kb, hid, (int)rows, hid, hid, s);  // proj -> Kb
  size_t tot = rows * (size_t)hid;
  gelu_mul_kernel<<<(tot + 255) / 256, 256, 0, s>>>(Hb, Kb, Qb, rows, hid);
  gemm(Qb, hid, p.f2w, hid, p.f2b, R, C, (int)rows, C, hid, s);
  size_t tt = (size_t)Bn * C * N;
  resid_spatial_kernel<<<(tt + 255) / 256, 256, 0, s>>>(R, x, y, Bn, C, N);
}

static void gated_spectral(const float* x, float* y, int Bn, int C, int H, int W,
                           int hid, const GP& p, float* arena, hipStream_t s)
{
  int F = H * W;                 // feature dim (LN/fc dims)
  size_t rows = (size_t)Bn * C;  // token rows
  float* R  = arena;
  float* Hb = R + rows * F;
  float* Qb = Hb + rows * (size_t)(2 * hid);
  float* Kb = Qb + rows * hid;
  float* Vb = Kb + rows * hid;
  float* Ob = Vb + rows * hid;

  ln_spectral_kernel<<<(rows * 32 + 255) / 256, 256, 0, s>>>(
      x, p.ng, p.nb, R, Bn, C, H, W);
  gemm(R, F, p.f1w, F, p.f1b, Hb, 2 * hid, (int)rows, 2 * hid, F, s);
  gemm(Hb + hid, 2 * hid, p.qw, hid, p.qb, Qb, hid, (int)rows, hid, hid, s);
  gemm(Hb + hid, 2 * hid, p.kw, hid, p.kb, Kb, hid, (int)rows, hid, hid, s);
  gemm(Hb + hid, 2 * hid, p.vw, hid, p.vb, Vb, hid, (int)rows, hid, hid, s);
  std_attn_kernel<<<Bn, 128, 0, s>>>(Qb, Kb, Vb, Ob, C, hid);
  gemm(Ob, hid, p.pw, hid, p.pb, Kb, hid, (int)rows, hid, hid, s);  // proj
  size_t tot = rows * (size_t)hid;
  gelu_mul_kernel<<<(tot + 255) / 256, 256, 0, s>>>(Hb, Kb, Qb, rows, hid);
  gemm(Qb, hid, p.f2w, hid, p.f2b, R, F, (int)rows, F, hid, s);
  size_t tt = (size_t)Bn * C * F;
  resid_spectral_kernel<<<(tt + 255) / 256, 256, 0, s>>>(R, x, y, Bn, C, H, W);
}

extern "C" void kernel_launch(void* const* d_in, const int* in_sizes, int n_in,
                              void* d_out, int out_size, void* d_ws, size_t ws_size,
                              hipStream_t stream)
{
  (void)in_sizes; (void)n_in; (void)out_size; (void)ws_size;

  const float* x1 = (const float*)d_in[0];
  const float* x2 = (const float*)d_in[1];
  // sorted-key pytree flatten: block1[2..15], block2[16..29], conv1_b[30],
  // conv1_w[31], conv2_b[32], conv2_w[33], layer1[34..47], layer2[48..61]
  GP b1p = mkGP(d_in, 2);
  GP b2p = mkGP(d_in, 16);
  const float* c1b = (const float*)d_in[30];
  const float* c1w = (const float*)d_in[31];
  const float* c2b = (const float*)d_in[32];
  const float* c2w = (const float*)d_in[33];
  GP l1p = mkGP(d_in, 34);
  GP l2p = mkGP(d_in, 48);

  const int B = 2048, Cin = 200;
  float* ws = (float*)d_ws;
  size_t off = 0;
  auto carve = [&](size_t n) { float* r = ws + off; off += n; return r; };

  const size_t S1 = (size_t)B * 64 * 49;   // stage-1 NCHW tensors
  const size_t S2 = (size_t)B * 128 * 25;  // stage-2 NCHW tensors
  float* c1  = carve(S1);
  float* c2  = carve(S1);
  float* a1o = carve(S1);
  float* a2o = carve(S1);
  float* b1o = carve(S1);
  float* b2o = carve(S1);
  float* ac1 = carve(S2);
  float* ac2 = carve(S2);
  float* bc1 = carve(S2);
  float* bc2 = carve(S2);
  float* arena = ws + off;  // shared scratch, reused per gated-block call

  // ---- conv1 (200 -> 64, 3x3, pad 1) on both streams
  {
    int M = B * 49;
    dim3 g((M + 63) / 64, (64 + 15) / 16);
    conv3x3_wmma_kernel<7, 7, 7, 7, 1><<<g, 128, 0, stream>>>(
        x1, c1w, c1b, c1, B, Cin, 64);
    conv3x3_wmma_kernel<7, 7, 7, 7, 1><<<g, 128, 0, stream>>>(
        x2, c1w, c1b, c2, B, Cin, 64);
  }
  // ---- fused spatial-then-channel exchange (in place)
  exchange_kernel<<<(S1 + 255) / 256, 256, 0, stream>>>(c1, c2, S1, 64, 7, 7, 1, 1);

  // ---- stage-1 gated blocks
  gated_spatial (c1, a1o, B, 64, 7, 7, 170, b1p, arena, stream);
  gated_spatial (c2, a2o, B, 64, 7, 7, 170, b1p, arena, stream);
  gated_spectral(c1, b1o, B, 64, 7, 7, 130, l1p, arena, stream);
  gated_spectral(c2, b2o, B, 64, 7, 7, 130, l1p, arena, stream);

  // ---- conv2 (64 -> 128, 3x3, pad 0): 7x7 -> 5x5 on all four tensors
  {
    int M = B * 25;
    dim3 g((M + 63) / 64, (128 + 15) / 16);
    conv3x3_wmma_kernel<7, 7, 5, 5, 0><<<g, 128, 0, stream>>>(
        a1o, c2w, c2b, ac1, B, 64, 128);
    conv3x3_wmma_kernel<7, 7, 5, 5, 0><<<g, 128, 0, stream>>>(
        a2o, c2w, c2b, ac2, B, 64, 128);
    conv3x3_wmma_kernel<7, 7, 5, 5, 0><<<g, 128, 0, stream>>>(
        b1o, c2w, c2b, bc1, B, 64, 128);
    conv3x3_wmma_kernel<7, 7, 5, 5, 0><<<g, 128, 0, stream>>>(
        b2o, c2w, c2b, bc2, B, 64, 128);
  }
  exchange_kernel<<<(S2 + 255) / 256, 256, 0, stream>>>(ac1, ac2, S2, 128, 5, 5, 1, 0);
  exchange_kernel<<<(S2 + 255) / 256, 256, 0, stream>>>(bc1, bc2, S2, 128, 5, 5, 0, 1);

  // ---- stage-2 gated blocks write straight into the stacked output
  float* out = (float*)d_out;
  const size_t Q = S2;
  gated_spatial (ac1, out + 0 * Q, B, 128, 5, 5, 341, b2p, arena, stream);
  gated_spatial (ac2, out + 1 * Q, B, 128, 5, 5, 341, b2p, arena, stream);
  gated_spectral(bc1, out + 2 * Q, B, 128, 5, 5, 66,  l2p, arena, stream);
  gated_spectral(bc2, out + 3 * Q, B, 128, 5, 5, 66,  l2p, arena, stream);
}
